// matching_net_nlp_86577950753069
// MI455X (gfx1250) — matrix operand order repaired and checked by reading, not compiled, not run on hardware
//
#include <hip/hip_runtime.h>
#include <hip/hip_bf16.h>

typedef __attribute__((ext_vector_type(16))) __bf16 v16bf;
typedef __attribute__((ext_vector_type(8)))  float  v8f;

#define VOCAB 13000
#define D     300
#define DP    320      // D padded to multiple of 32
#define M     500
#define MP    512      // M padded to multiple of 32
#define BEP   64
#define NW    5
#define T     128
#define NSEQ  384      // 320 support + 64 target
#define NST   24       // sequence tiles of 16
#define KT_U  16       // MP/32 k-tiles for recurrent GEMM
#define KT_W  10       // DP/32 k-tiles for input projection
#define NMT   32       // MP/16 m-tiles per gate
#define NT_O  19       // ceil(300/16) n-tiles for output GEMM
#define KT_O  32       // 1024/32 k-tiles for output GEMM
#define EOUT_LD 304

// ---------------------------------------------------------------------------
// WMMA fragment layout helpers (CDNA5, wave32).
// A (16xK, bf16): lane l -> row m = l%16; K per v16bf element e:
//   k = kt*32 + (l/16)*16 + e
// B (Kx16, bf16): lane l -> col n = l%16; k = kt*32 + (l/16)*16 + e
// C/D (16x16, f32): vgpr j -> row m = j + 8*(l/16), col n = l%16
// ---------------------------------------------------------------------------
__device__ __forceinline__ unsigned a_elem_addr(unsigned s, unsigned k) {
  unsigned kt = k >> 5, kl = k & 31;
  unsigned lane, e;
  lane = s + ((kl >> 4) << 4); e = kl & 15u;
  return kt * 512u + lane * 16u + e;
}

// CDNA5 async global->LDS copy (ASYNCcnt-tracked, bypasses VGPRs).
__device__ __forceinline__ void async_g2lds_b128(unsigned lds_off, const float* g) {
  asm volatile("global_load_async_to_lds_b128 %0, %1, off"
               :: "v"(lds_off), "v"(g) : "memory");
}
__device__ __forceinline__ void wait_async0() {
  asm volatile("s_wait_asynccnt 0x0" ::: "memory");
}

// ---------------------------------------------------------------------------
// Weight packing kernels: f32 -> bf16 in B-fragment order.
// ---------------------------------------------------------------------------
__global__ void pack_u_kernel(const float* __restrict__ UL,
                              const float* __restrict__ UR,
                              __bf16* __restrict__ UB) {
  // UB[dir][gate][mt][kt][lane*16+e], total 2*3*32*16*512 = 2*3*2^18
  size_t idx = (size_t)blockIdx.x * blockDim.x + threadIdx.x;
  size_t total = (size_t)2 * 3 * NMT * KT_U * 512;
  if (idx >= total) return;
  unsigned e    = idx & 15u;
  unsigned lane = (idx >> 4) & 31u;
  unsigned kt   = (idx >> 9) & 15u;
  unsigned mt   = (idx >> 13) & 31u;
  unsigned g    = (unsigned)((idx >> 18) % 3u);
  unsigned dir  = (unsigned)(idx / ((size_t)3 << 18));
  unsigned m = mt * 16u + (lane & 15u);
  unsigned k = kt * 32u + (lane >> 4) * 16u + e;
  float v = 0.f;
  if (m < M && k < M) {
    const float* U = dir ? UR : UL;
    v = U[((size_t)g * M + m) * M + k];
  }
  UB[idx] = (__bf16)v;
}

__global__ void pack_w_kernel(const float* __restrict__ WL,
                              const float* __restrict__ WR,
                              __bf16* __restrict__ WB) {
  // WB[dir][gate][mt][kt][lane*16+e], total 2*3*32*10*512 = 983040
  size_t idx = (size_t)blockIdx.x * blockDim.x + threadIdx.x;
  size_t total = (size_t)2 * 3 * NMT * KT_W * 512;
  if (idx >= total) return;
  unsigned e    = (unsigned)(idx % 16u);
  unsigned lane = (unsigned)((idx / 16u) % 32u);
  unsigned kt   = (unsigned)((idx / 512u) % KT_W);
  unsigned mt   = (unsigned)((idx / (512u * KT_W)) % NMT);
  unsigned g    = (unsigned)((idx / (512u * KT_W * NMT)) % 3u);
  unsigned dir  = (unsigned)(idx / ((size_t)512 * KT_W * NMT * 3));
  unsigned m = mt * 16u + (lane & 15u);
  unsigned k = kt * 32u + (lane >> 4) * 16u + e;   // k = embedding dim d
  float v = 0.f;
  if (m < M && k < D) {
    const float* W = dir ? WR : WL;
    v = W[((size_t)g * M + m) * D + k];
  }
  WB[idx] = (__bf16)v;
}

__global__ void pack_o_kernel(const float* __restrict__ Wo,
                              const float* __restrict__ Uo,
                              __bf16* __restrict__ OB) {
  // OB[nt][kt][lane*16+e], K = [l(0..511) ; r(512..1023)], total 19*32*512
  size_t idx = (size_t)blockIdx.x * blockDim.x + threadIdx.x;
  size_t total = (size_t)NT_O * KT_O * 512;
  if (idx >= total) return;
  unsigned e    = (unsigned)(idx % 16u);
  unsigned lane = (unsigned)((idx / 16u) % 32u);
  unsigned kt   = (unsigned)((idx / 512u) % KT_O);
  unsigned nt   = (unsigned)(idx / (512u * KT_O));
  unsigned n = nt * 16u + (lane & 15u);           // output dim d
  unsigned k = kt * 32u + (lane >> 4) * 16u + e;  // combined K
  float v = 0.f;
  if (n < D) {
    if (k < MP)           { unsigned m = k;       if (m < M) v = Wo[(size_t)n * M + m]; }
    else                  { unsigned m = k - MP;  if (m < M) v = Uo[(size_t)n * M + m]; }
  }
  OB[idx] = (__bf16)v;
}

// ---------------------------------------------------------------------------
// Bidirectional recurrent scan. grid = (24 seq-tiles, 2 directions),
// block = 256 (8 waves). Each wave owns 4 m-tiles (16 cells each) and
// computes all 3 gate tiles for them; f32 cell state lives in registers
// (creg[4][8]) since the (s,m) partition is identical every timestep.
// ---------------------------------------------------------------------------
__global__ __launch_bounds__(256) void recurrence_kernel(
    const int* __restrict__ sup_tok, const int* __restrict__ tgt_tok,
    const int* __restrict__ sup_blank, const int* __restrict__ tgt_blank,
    const float* __restrict__ emb,
    const __bf16* __restrict__ UB, const __bf16* __restrict__ WB,
    const float* __restrict__ bL, const float* __restrict__ bR,
    const float* __restrict__ c0L, const float* __restrict__ c0R,
    float* __restrict__ env /* [2][NSEQ][MP] */)
{
  __shared__ __align__(16) float xstage[16 * D];   // 18.75 KB async landing pad
  __shared__ __bf16 cA[KT_U * 512];                // 16 KB, c in A-fragment order
  __shared__ __bf16 xA[KT_W * 512];                // 10 KB, x in A-fragment order
  __shared__ int    tokb[16];
  __shared__ int    snapb[16];

  const unsigned st  = blockIdx.x;
  const unsigned dir = blockIdx.y;       // 0 = L, 1 = R (reversed sequence)
  const unsigned tid = threadIdx.x;
  const unsigned l   = tid & 31u;
  const unsigned wv  = tid >> 5;
  const float* bvec = dir ? bR : bL;
  const float* c0   = dir ? c0R : c0L;

  // register-resident f32 cell state for this thread's cells
  float creg[4][8];
  unsigned mq[4];
  #pragma unroll
  for (unsigned q = 0; q < 4u; ++q) {
    mq[q] = (wv * 4u + q) * 16u + (l & 15u);
    #pragma unroll
    for (int j = 0; j < 8; ++j)
      creg[q][j] = (mq[q] < M) ? c0[mq[q]] : 0.f;
  }
  if (tid < 16u) {
    unsigned seq = st * 16u + tid;
    int bp = (seq < 320u) ? sup_blank[seq] : tgt_blank[seq - 320u];
    snapb[tid] = dir ? (T - 1 - bp) : bp;   // step at which pre-state is saved
  }
  __syncthreads();

  for (int t = 0; t < T; ++t) {
    const int tt = dir ? (T - 1 - t) : t;
    if (tid < 16u) {
      unsigned seq = st * 16u + tid;
      tokb[tid] = (seq < 320u) ? sup_tok[(size_t)seq * T + tt]
                               : tgt_tok[(size_t)(seq - 320u) * T + tt];
    }
    // scatter current cell state (registers) -> bf16 A fragments
    #pragma unroll
    for (unsigned q = 0; q < 4u; ++q) {
      #pragma unroll
      for (int j = 0; j < 8; ++j) {
        unsigned s = (unsigned)j + ((l >> 4) << 3);
        cA[a_elem_addr(s, mq[q])] = (__bf16)creg[q][j];
      }
    }
    __syncthreads();   // tokb + cA ready; prev-iter WMMA reads fenced below

    // async-gather this step's 16 embedding rows straight into LDS
    for (unsigned idx = tid; idx < 16u * (D / 4); idx += 256u) {
      unsigned s  = idx / (D / 4);
      unsigned c4 = (idx % (D / 4)) * 4u;
      unsigned lds_off = (unsigned)(uintptr_t)(&xstage[s * D + c4]);
      async_g2lds_b128(lds_off, emb + (size_t)tokb[s] * D + c4);
    }
    wait_async0();
    __syncthreads();   // xstage populated for all waves

    // convert staged f32 rows -> bf16 A fragments (with DP zero padding)
    for (unsigned i = tid; i < 16u * DP; i += 256u) {
      unsigned s = i / DP, d = i % DP;
      float xv = (d < D) ? xstage[s * D + d] : 0.f;
      xA[a_elem_addr(s, d)] = (__bf16)xv;
    }
    __syncthreads();   // xA ready

    #pragma unroll
    for (unsigned q = 0; q < 4u; ++q) {
      const unsigned mt = wv * 4u + q;
      const unsigned m  = mq[q];

      v8f acc[3];
      #pragma unroll
      for (int g = 0; g < 3; ++g) {
        float bb = (m < M) ? bvec[g * M + m] : 0.f;
        acc[g] = (v8f){bb, bb, bb, bb, bb, bb, bb, bb};
      }
      // hint the next m-tile's weights toward the WGP while we grind this one
      if (mt + 1u < NMT)
        __builtin_prefetch(UB + ((((size_t)dir * 3) * NMT + mt + 1) * KT_U) * 512 + l * 16, 0, 1);

      #pragma unroll
      for (unsigned kt = 0; kt < KT_U; ++kt) {
        v16bf a = *(const v16bf*)&cA[kt * 512u + l * 16u];
        #pragma unroll
        for (int g = 0; g < 3; ++g) {
          v16bf b = *(const v16bf*)(UB + ((((size_t)dir * 3 + g) * NMT + mt) * KT_U + kt) * 512 + l * 16);
          acc[g] = __builtin_amdgcn_wmma_f32_16x16x32_bf16(false, a, false, b,
                                                           (short)0, acc[g], false, false);
        }
      }
      #pragma unroll
      for (unsigned kt = 0; kt < KT_W; ++kt) {
        v16bf a = *(const v16bf*)&xA[kt * 512u + l * 16u];
        #pragma unroll
        for (int g = 0; g < 3; ++g) {
          v16bf b = *(const v16bf*)(WB + ((((size_t)dir * 3 + g) * NMT + mt) * KT_W + kt) * 512 + l * 16);
          acc[g] = __builtin_amdgcn_wmma_f32_16x16x32_bf16(false, a, false, b,
                                                           (short)0, acc[g], false, false);
        }
      }

      // gates + cell update entirely in registers
      #pragma unroll
      for (int j = 0; j < 8; ++j) {
        unsigned s = (unsigned)j + ((l >> 4) << 3);
        float f  = 1.f / (1.f + __expf(-acc[0][j]));
        float ii = 1.f / (1.f + __expf(-acc[1][j]));
        float cb = tanhf(acc[2][j]);
        float cold = creg[q][j];
        if (t == snapb[s])
          env[((size_t)dir * NSEQ + st * 16u + s) * MP + m] = cold;  // pre-state
        creg[q][j] = (m < M) ? (f * cold + ii * cb) : 0.f;
      }
    }
    __syncthreads();   // all waves done reading cA before next-iter scatter
  }
}

// ---------------------------------------------------------------------------
// e = tanh(l@Wo^T + r@Uo^T + bo) : [384,1024] x [1024,304] WMMA GEMM.
// ---------------------------------------------------------------------------
__global__ __launch_bounds__(128) void outenv_kernel(
    const float* __restrict__ env, const __bf16* __restrict__ OB,
    const float* __restrict__ bo, float* __restrict__ eout)
{
  __shared__ __bf16 aO[KT_O * 512];   // 32 KB: [l ; r] A fragments, K = 1024
  const unsigned st  = blockIdx.x;
  const unsigned tid = threadIdx.x;
  const unsigned l   = tid & 31u;
  const unsigned wv  = tid >> 5;

  for (unsigned i = tid; i < 16u * 1024u; i += 128u) {
    unsigned s = i >> 10, k = i & 1023u;
    unsigned half = k >> 9, mm = k & 511u;
    float v = env[((size_t)half * NSEQ + st * 16u + s) * MP + mm];
    aO[a_elem_addr(s, k)] = (__bf16)v;
  }
  __syncthreads();

  for (unsigned nt = wv; nt < NT_O; nt += 4u) {
    unsigned n = nt * 16u + (l & 15u);
    float bb = (n < D) ? bo[n] : 0.f;
    v8f acc = (v8f){bb, bb, bb, bb, bb, bb, bb, bb};
    #pragma unroll
    for (unsigned kt = 0; kt < KT_O; ++kt) {
      v16bf a = *(const v16bf*)&aO[kt * 512u + l * 16u];
      v16bf b = *(const v16bf*)(OB + ((size_t)(nt * KT_O + kt)) * 512 + l * 16);
      acc = __builtin_amdgcn_wmma_f32_16x16x32_bf16(false, a, false, b,
                                                    (short)0, acc, false, false);
    }
    if (n < D) {
      #pragma unroll
      for (int j = 0; j < 8; ++j) {
        unsigned s = (unsigned)j + ((l >> 4) << 3);
        eout[(size_t)(st * 16u + s) * EOUT_LD + n] = tanhf(acc[j]);
      }
    }
  }
}

// ---------------------------------------------------------------------------
// Per-episode cosine similarity, double softmax, accuracy + CE loss.
// ---------------------------------------------------------------------------
__global__ void classify_kernel(const float* __restrict__ eout,
                                const int* __restrict__ target_y,
                                float* __restrict__ ep)
{
  __shared__ float sdata[64];
  __shared__ float res[11];
  const unsigned b = blockIdx.x, tid = threadIdx.x;
  const float* tg = eout + (size_t)(320 + b) * EOUT_LD;

  for (int r = 0; r < 11; ++r) {
    float p = 0.f;
    if (r < 5) {
      const float* sp = eout + (size_t)(b * NW + r) * EOUT_LD;
      for (unsigned d = tid; d < D; d += 64u) p += sp[d] * tg[d];
    } else if (r < 10) {
      const float* sp = eout + (size_t)(b * NW + (r - 5)) * EOUT_LD;
      for (unsigned d = tid; d < D; d += 64u) p += sp[d] * sp[d];
    } else {
      for (unsigned d = tid; d < D; d += 64u) p += tg[d] * tg[d];
    }
    sdata[tid] = p; __syncthreads();
    for (unsigned off = 32u; off > 0u; off >>= 1) {
      if (tid < off) sdata[tid] += sdata[tid + off];
      __syncthreads();
    }
    if (tid == 0) res[r] = sdata[0];
    __syncthreads();
  }

  if (tid == 0) {
    float ntg = sqrtf(res[10]);
    float z[NW], zmax = -1e30f;
    for (int w = 0; w < NW; ++w) {
      float den = fmaxf(sqrtf(res[5 + w]) * ntg, 1e-8f);
      z[w] = (res[w] / den) * 10.f;
      zmax = fmaxf(zmax, z[w]);
    }
    float se = 0.f, p[NW];
    for (int w = 0; w < NW; ++w) { p[w] = __expf(z[w] - zmax); se += p[w]; }
    int amax = 0; float pmax = -1.f;
    for (int w = 0; w < NW; ++w) {
      p[w] /= se;
      if (p[w] > pmax) { pmax = p[w]; amax = w; }
    }
    // cross-entropy on the softmax probs (double softmax, as in reference)
    float m2 = -1e30f;
    for (int w = 0; w < NW; ++w) m2 = fmaxf(m2, p[w]);
    float s2 = 0.f;
    for (int w = 0; w < NW; ++w) s2 += __expf(p[w] - m2);
    int y = target_y[b];
    float logp = (p[y] - m2) - __logf(s2);
    ep[b * 2 + 0] = (amax == y) ? 1.f : 0.f;
    ep[b * 2 + 1] = -logp;
  }
}

__global__ void finalize_kernel(const float* __restrict__ ep, float* __restrict__ out) {
  __shared__ float sa[64], sl[64];
  unsigned tid = threadIdx.x;
  sa[tid] = ep[tid * 2];
  sl[tid] = ep[tid * 2 + 1];
  __syncthreads();
  for (unsigned off = 32u; off > 0u; off >>= 1) {
    if (tid < off) { sa[tid] += sa[tid + off]; sl[tid] += sl[tid + off]; }
    __syncthreads();
  }
  if (tid == 0) { out[0] = sa[0] / (float)BEP; out[1] = sl[0] / (float)BEP; }
}

// ---------------------------------------------------------------------------
extern "C" void kernel_launch(void* const* d_in, const int* in_sizes, int n_in,
                              void* d_out, int out_size, void* d_ws, size_t ws_size,
                              hipStream_t stream) {
  (void)in_sizes; (void)n_in; (void)out_size; (void)ws_size;
  const int*   sup_tok   = (const int*)d_in[0];
  const int*   tgt_tok   = (const int*)d_in[1];
  const int*   sup_blank = (const int*)d_in[2];
  const int*   tgt_blank = (const int*)d_in[3];
  const int*   target_y  = (const int*)d_in[4];
  const float* emb       = (const float*)d_in[5];
  const float* WL        = (const float*)d_in[6];
  const float* UL        = (const float*)d_in[7];
  const float* bL        = (const float*)d_in[8];
  const float* WR        = (const float*)d_in[9];
  const float* UR        = (const float*)d_in[10];
  const float* bR        = (const float*)d_in[11];
  const float* Wo        = (const float*)d_in[12];
  const float* Uo        = (const float*)d_in[13];
  const float* bo        = (const float*)d_in[14];
  const float* c0L       = (const float*)d_in[15];
  const float* c0R       = (const float*)d_in[16];

  char* ws = (char*)d_ws;
  size_t off = 0;
  __bf16* UB  = (__bf16*)(ws + off); off += (size_t)2 * 3 * NMT * KT_U * 512 * 2;
  __bf16* WB  = (__bf16*)(ws + off); off += (size_t)2 * 3 * NMT * KT_W * 512 * 2;
  __bf16* OB  = (__bf16*)(ws + off); off += (size_t)NT_O * KT_O * 512 * 2;
  float*  env = (float*)(ws + off);  off += (size_t)2 * NSEQ * MP * 4;
  float*  eout= (float*)(ws + off);  off += (size_t)NSEQ * EOUT_LD * 4;
  float*  ep  = (float*)(ws + off);  off += (size_t)BEP * 2 * 4;

  {
    int n = 2 * 3 * NMT * KT_U * 512;
    pack_u_kernel<<<(n + 255) / 256, 256, 0, stream>>>(UL, UR, UB);
  }
  {
    int n = 2 * 3 * NMT * KT_W * 512;
    pack_w_kernel<<<(n + 255) / 256, 256, 0, stream>>>(WL, WR, WB);
  }
  {
    int n = NT_O * KT_O * 512;
    pack_o_kernel<<<(n + 255) / 256, 256, 0, stream>>>(Wo, Uo, OB);
  }
  recurrence_kernel<<<dim3(NST, 2), 256, 0, stream>>>(
      sup_tok, tgt_tok, sup_blank, tgt_blank, emb, UB, WB, bL, bR, c0L, c0R, env);
  outenv_kernel<<<NST, 128, 0, stream>>>(env, OB, bo, eout);
  classify_kernel<<<BEP, 64, 0, stream>>>(eout, target_y, ep);
  finalize_kernel<<<1, 64, 0, stream>>>(ep, (float*)d_out);
}
